// MSARowAttentionWithPairBias_20856361189897
// MI455X (gfx1250) — matrix operand order repaired and checked by reading, not compiled, not run on hardware
//
#include <hip/hip_runtime.h>

// ---------------------------------------------------------------------------
// MSA Row Attention With Pair Bias — MI455X (gfx1250, wave32, WMMA bf16)
// B=1, S=128, L=256, D=256, H=8, C=32.  All GEMM-shaped math runs on
// v_wmma_f32_16x16x32_bf16; global->LDS staging uses the CDNA5 async path
// (GLOBAL_LOAD_ASYNC_TO_LDS_B128 + s_wait_asynccnt) when available.
// ---------------------------------------------------------------------------

typedef __bf16 bf16;
typedef __attribute__((ext_vector_type(8)))  bf16  bf16x8;
typedef __attribute__((ext_vector_type(16))) bf16  bf16x16;
typedef __attribute__((ext_vector_type(8)))  float v8f;
typedef int v4i __attribute__((vector_size(16)));   // matches builtin param type

#define NTHREADS 256
#define S_DIM 128
#define L_DIM 256
#define D_DIM 256
#define H_DIM 8
#define C_DIM 32
#define M_TOTAL (S_DIM * L_DIM)          // 32768 rows
#define NEG_INF_F (-10000.0f)

#define AS1 __attribute__((address_space(1)))
#define AS3 __attribute__((address_space(3)))

#if defined(__has_builtin)
#  if __has_builtin(__builtin_amdgcn_global_load_async_to_lds_b128)
#    define HAVE_ASYNC_LDS 1
#  endif
#endif
#ifndef HAVE_ASYNC_LDS
#  define HAVE_ASYNC_LDS 0
#endif

// 16-byte global -> LDS copy. Async (ASYNCcnt-tracked) when the gfx1250
// builtin exists; synchronous VGPR round-trip otherwise.
__device__ __forceinline__ void copy16_g2l(bf16* dst_lds, const bf16* src_glob) {
#if HAVE_ASYNC_LDS
    __builtin_amdgcn_global_load_async_to_lds_b128(
        (AS1 v4i*)(unsigned long long)src_glob,
        (AS3 v4i*)(unsigned int)(unsigned long long)dst_lds,
        0, 0);
#else
    *(bf16x8*)dst_lds = *(const bf16x8*)src_glob;
#endif
}

// Wait for outstanding async LDS copies, then workgroup barrier.
__device__ __forceinline__ void async_join_barrier() {
#if HAVE_ASYNC_LDS
#  if __has_builtin(__builtin_amdgcn_s_wait_asynccnt)
    __builtin_amdgcn_s_wait_asynccnt(0);
#  else
    asm volatile("s_wait_asynccnt 0x0" ::: "memory");
#  endif
#endif
    __syncthreads();
}

__device__ __forceinline__ v8f vzero8() {
    v8f z = {0.f, 0.f, 0.f, 0.f, 0.f, 0.f, 0.f, 0.f};
    return z;
}

__device__ __forceinline__ bf16x16 join8(bf16x8 a, bf16x8 b) {
    return __builtin_shufflevector(a, b, 0,1,2,3,4,5,6,7,8,9,10,11,12,13,14,15);
}

__device__ __forceinline__ v8f wmma_bf16(bf16x16 a, bf16x16 b, v8f c) {
    // D = A(16x32) * B(32x16) + C, fp32 accumulate
    return __builtin_amdgcn_wmma_f32_16x16x32_bf16(
        /*neg_a=*/false, a, /*neg_b=*/false, b,
        /*c_mod=*/(short)0, c, /*reuse_a=*/false, /*reuse_b=*/false);
}

// A-fragment (16x32, row-major source with given row stride).
// ISA layout: lane l<16 holds row l, K in {kb..kb+7} U {16+kb..16+kb+7}, kb=(l>>4)*8
__device__ __forceinline__ bf16x16 load_a_frag(const bf16* base, int stride, int lane) {
    int r  = lane & 15;
    int kb = (lane >> 4) * 8;
    const bf16* p = base + r * stride;
    bf16x8 lo = *(const bf16x8*)(p + kb);
    bf16x8 hi = *(const bf16x8*)(p + 16 + kb);
    return join8(lo, hi);
}

// B-fragment (32x16) from a [col][k] layout with given col stride.
// ISA layout: lane l holds col (l&15), K in {kb..kb+7} U {16+kb..16+kb+7}, kb=(l>>4)*8
__device__ __forceinline__ bf16x16 load_b_frag(const bf16* base, int stride, int lane) {
    int c  = lane & 15;
    int kb = (lane >> 4) * 8;
    const bf16* p = base + c * stride + kb;
    bf16x8 lo = *(const bf16x8*)(p);
    bf16x8 hi = *(const bf16x8*)(p + 16);
    return join8(lo, hi);
}

// Cooperative copy of a 64x32 bf16 tile (row-major src) into LDS [64][32].
__device__ __forceinline__ void load_tile_64x32(bf16* dst, const bf16* src,
                                                int src_stride, int tid) {
    int r  = tid >> 2;        // 0..63
    int cb = (tid & 3) * 8;   // 0,8,16,24
    copy16_g2l(dst + r * 32 + cb, src + (size_t)r * src_stride + cb);
}

// ---------------------------------------------------------------------------
// Kernel 1: LayerNorm over D, fp32 -> bf16. One wave per row (wave32).
// ---------------------------------------------------------------------------
__global__ void msa_ln_kernel(const float* __restrict__ m,
                              const float* __restrict__ ln_g,
                              const float* __restrict__ ln_b,
                              bf16* __restrict__ mn) {
    int lane = threadIdx.x & 31;
    int wid  = threadIdx.x >> 5;
    int row  = blockIdx.x * 8 + wid;

    const float* p = m + (size_t)row * D_DIM + lane * 8;
    float4 v0 = ((const float4*)p)[0];
    float4 v1 = ((const float4*)p)[1];
    float x[8] = {v0.x, v0.y, v0.z, v0.w, v1.x, v1.y, v1.z, v1.w};

    float sum = 0.f, sq = 0.f;
    #pragma unroll
    for (int j = 0; j < 8; ++j) { sum += x[j]; sq += x[j] * x[j]; }
    #pragma unroll
    for (int off = 16; off > 0; off >>= 1) {
        sum += __shfl_xor(sum, off, 32);
        sq  += __shfl_xor(sq,  off, 32);
    }
    float mean = sum * (1.0f / D_DIM);
    float var  = sq * (1.0f / D_DIM) - mean * mean;
    float rinv = rsqrtf(var + 1e-5f);

    bf16x8 o;
    #pragma unroll
    for (int j = 0; j < 8; ++j) {
        int d = lane * 8 + j;
        o[j] = (bf16)((x[j] - mean) * rinv * ln_g[d] + ln_b[d]);
    }
    *(bf16x8*)(mn + (size_t)row * D_DIM + lane * 8) = o;
}

// ---------------------------------------------------------------------------
// Kernel 2: weights fp32 -> bf16 (Wq,Wk,Wv,Wg,Wo packed consecutively).
// ---------------------------------------------------------------------------
__global__ void msa_cvtw_kernel(const float* __restrict__ w0, const float* __restrict__ w1,
                                const float* __restrict__ w2, const float* __restrict__ w3,
                                const float* __restrict__ w4, bf16* __restrict__ dst) {
    int i = blockIdx.x * blockDim.x + threadIdx.x;   // 0 .. 5*65536-1
    const float* srcs[5] = {w0, w1, w2, w3, w4};
    int which = i >> 16;
    int off   = i & 65535;
    dst[i] = (bf16)srcs[which][off];
}

// ---------------------------------------------------------------------------
// Kernel 3: fused Q/K/V/G projection.  out = mn @ W^T, 4 weights share A-tile.
// Grid (M/64, D/64), 256 threads = 8 waves. Wave: 32 rows x 16 cols x 4 outs.
// Q scaled by 1/sqrt(C); G gets +bg then sigmoid; V stored transposed (S,H,C,L).
// ---------------------------------------------------------------------------
__global__ void msa_qkvg_kernel(const bf16* __restrict__ mn,
                                const bf16* __restrict__ wq, const bf16* __restrict__ wk,
                                const bf16* __restrict__ wv, const bf16* __restrict__ wg,
                                const float* __restrict__ bg,
                                bf16* __restrict__ q_ws, bf16* __restrict__ k_ws,
                                bf16* __restrict__ v_ws, bf16* __restrict__ g_ws) {
    __shared__ bf16 aTile[64 * 32];
    __shared__ bf16 bTile[4][64 * 32];

    int tid  = threadIdx.x;
    int lane = tid & 31;
    int wid  = tid >> 5;
    int blockM = blockIdx.x * 64;
    int blockN = blockIdx.y * 64;
    int subM = (wid >> 2) * 32;   // 0 or 32
    int subN = (wid & 3) * 16;    // 0,16,32,48

    const bf16* wsrc[4] = {wq, wk, wv, wg};

    v8f acc[2][4];
    #pragma unroll
    for (int h2 = 0; h2 < 2; ++h2)
        #pragma unroll
        for (int wi = 0; wi < 4; ++wi) acc[h2][wi] = vzero8();

    for (int kb = 0; kb < D_DIM; kb += 32) {
        load_tile_64x32(aTile, mn + (size_t)blockM * D_DIM + kb, D_DIM, tid);
        #pragma unroll
        for (int wi = 0; wi < 4; ++wi)
            load_tile_64x32(bTile[wi], wsrc[wi] + (size_t)blockN * D_DIM + kb, D_DIM, tid);
        if (kb + 32 < D_DIM)
            __builtin_prefetch(mn + (size_t)blockM * D_DIM + kb + 32, 0, 0);
        async_join_barrier();

        bf16x16 aLo = load_a_frag(aTile + subM * 32, 32, lane);
        bf16x16 aHi = load_a_frag(aTile + (subM + 16) * 32, 32, lane);
        #pragma unroll
        for (int wi = 0; wi < 4; ++wi) {
            bf16x16 bF = load_b_frag(bTile[wi] + subN * 32, 32, lane);
            acc[0][wi] = wmma_bf16(aLo, bF, acc[0][wi]);
            acc[1][wi] = wmma_bf16(aHi, bF, acc[1][wi]);
        }
        __syncthreads();
    }

    // Epilogue
    int nl = subN + (lane & 15);
    int n  = blockN + nl;
    int hh = n >> 5;          // head
    int cc = n & 31;          // channel in head
    float bgv = bg[n];
    const float qscale = 0.17677669529663687f;  // 1/sqrt(32)

    #pragma unroll
    for (int h2 = 0; h2 < 2; ++h2) {
        int mbase = blockM + subM + h2 * 16 + 8 * (lane >> 4);
        #pragma unroll
        for (int r = 0; r < 8; ++r) {
            int mrow = mbase + r;
            int s    = mrow >> 8;         // row / L
            int lres = mrow & 255;        // row % L
            size_t qi = ((size_t)(s * H_DIM + hh) * L_DIM + lres) * C_DIM + cc;
            q_ws[qi] = (bf16)(acc[h2][0][r] * qscale);
            k_ws[qi] = (bf16)(acc[h2][1][r]);
            v_ws[((size_t)(s * H_DIM + hh) * C_DIM + cc) * L_DIM + lres] = (bf16)(acc[h2][2][r]);
            float gv = acc[h2][3][r] + bgv;
            g_ws[qi] = (bf16)(1.0f / (1.0f + __expf(-gv)));
        }
    }
}

// ---------------------------------------------------------------------------
// Kernel 4: attention per (s,h).  Scores via WMMA (K=C=32, one step), softmax
// with deferred normalization, AV via WMMA (K=L=256, 8 steps).
// Grid = S*H = 1024 blocks, 256 threads. ~50KB LDS.
// ---------------------------------------------------------------------------
__global__ void msa_attn_kernel(const bf16* __restrict__ q_ws, const bf16* __restrict__ k_ws,
                                const bf16* __restrict__ v_ws, const bf16* __restrict__ g_ws,
                                const unsigned char* __restrict__ seq_pad,
                                const unsigned char* __restrict__ res_pad,
                                bf16* __restrict__ o_ws) {
    __shared__ bf16  kbuf[L_DIM * C_DIM];   // 16KB: keys [kk][c]
    __shared__ bf16  sbuf[64 * L_DIM];      // 32KB: score/prob chunk [64 q-rows][256 kk]
    __shared__ float pmax_s[4][64];
    __shared__ float psum_s[4][64];
    __shared__ float rowscale[64];

    int tid  = threadIdx.x;
    int lane = tid & 31;
    int wid  = tid >> 5;
    int sh   = blockIdx.x;                  // s*H + h
    int s    = sh >> 3;
    int h    = sh & 7;

    const bf16* qh = q_ws + (size_t)sh * L_DIM * C_DIM;
    const bf16* kh = k_ws + (size_t)sh * L_DIM * C_DIM;
    const bf16* vh = v_ws + (size_t)sh * C_DIM * L_DIM;
    const bf16* gh = g_ws + (size_t)sh * L_DIM * C_DIM;

    // Stage keys in LDS via async copies (layout matches global: [kk][c]).
    {
        const bf16* src = kh + tid * 32;
        bf16*       dst = kbuf + tid * 32;
        #pragma unroll
        for (int j = 0; j < 4; ++j)
            copy16_g2l(dst + j * 8, src + j * 8);
    }
    async_join_barrier();

    for (int qc = 0; qc < L_DIM; qc += 64) {
        // ---- scores: wave -> 16 q-rows x 128 kk-cols (8 tiles) ----
        int m0 = qc + (wid & 3) * 16;
        bf16x16 aq = load_a_frag(qh + (size_t)m0 * C_DIM, C_DIM, lane);
        int ncol0 = (wid >> 2) * 128;
        #pragma unroll
        for (int nt = 0; nt < 8; ++nt) {
            int n0 = ncol0 + nt * 16;
            bf16x16 bk = load_b_frag(kbuf + n0 * C_DIM, C_DIM, lane);
            v8f accS = wmma_bf16(aq, bk, vzero8());
            int ncol  = n0 + (lane & 15);
            int rb    = (m0 - qc) + 8 * (lane >> 4);
            #pragma unroll
            for (int r = 0; r < 8; ++r)
                sbuf[(rb + r) * L_DIM + ncol] = (bf16)accS[r];
        }
        __syncthreads();

        // ---- softmax: thread (seg,row) covers 64 cols of one row ----
        int row  = tid & 63;
        int seg  = tid >> 6;
        int qres = qc + row;
        float biasq = seq_pad[qres] ? NEG_INF_F : 0.0f;
        int col0 = seg * 64;

        float mx = -3.0e38f;
        for (int j = 0; j < 64; ++j) {
            int col = col0 + j;
            float x = (float)sbuf[row * L_DIM + col] + biasq
                      + (res_pad[col] ? NEG_INF_F : 0.0f);
            mx = fmaxf(mx, x);
        }
        pmax_s[seg][row] = mx;
        __syncthreads();

        float rmax = fmaxf(fmaxf(pmax_s[0][row], pmax_s[1][row]),
                           fmaxf(pmax_s[2][row], pmax_s[3][row]));
        float ssum = 0.f;
        for (int j = 0; j < 64; ++j) {
            int col = col0 + j;
            float x = (float)sbuf[row * L_DIM + col] + biasq
                      + (res_pad[col] ? NEG_INF_F : 0.0f);
            float e = __expf(x - rmax);
            sbuf[row * L_DIM + col] = (bf16)e;   // unnormalized prob
            ssum += e;
        }
        psum_s[seg][row] = ssum;
        __syncthreads();

        if (tid < 64) {
            float d = psum_s[0][tid] + psum_s[1][tid] + psum_s[2][tid] + psum_s[3][tid];
            rowscale[tid] = 1.0f / d;
        }
        __syncthreads();

        // ---- AV: wave -> 16 q-rows x 16 channels, K over 256 keys ----
        int prow = (wid & 3) * 16;       // row tile within chunk
        int cb0  = (wid >> 2) * 16;      // channel tile (0 or 16)
        v8f accO = vzero8();
        #pragma unroll
        for (int kt = 0; kt < 8; ++kt) {
            int kk0 = kt * 32;
            bf16x16 aP = load_a_frag(sbuf + prow * L_DIM + kk0, L_DIM, lane);
            bf16x16 bV = load_b_frag(vh + (size_t)cb0 * L_DIM + kk0, L_DIM, lane);
            accO = wmma_bf16(aP, bV, accO);
        }
        int c   = cb0 + (lane & 15);
        int rb2 = prow + 8 * (lane >> 4);
        #pragma unroll
        for (int r = 0; r < 8; ++r) {
            int rloc  = rb2 + r;
            int qres2 = qc + rloc;
            float val = accO[r] * rowscale[rloc] * (float)gh[(size_t)qres2 * C_DIM + c];
            o_ws[((size_t)(s * L_DIM + qres2)) * D_DIM + h * C_DIM + c] = (bf16)val;
        }
        __syncthreads();
    }
}

// ---------------------------------------------------------------------------
// Kernel 5: output projection out = o_ws @ Wo^T + bo (fp32 output).
// ---------------------------------------------------------------------------
__global__ void msa_oproj_kernel(const bf16* __restrict__ a,
                                 const bf16* __restrict__ wb,
                                 const float* __restrict__ bo,
                                 float* __restrict__ out) {
    __shared__ bf16 aTile[64 * 32];
    __shared__ bf16 bTile[64 * 32];

    int tid  = threadIdx.x;
    int lane = tid & 31;
    int wid  = tid >> 5;
    int blockM = blockIdx.x * 64;
    int blockN = blockIdx.y * 64;
    int subM = (wid >> 2) * 32;
    int subN = (wid & 3) * 16;

    v8f acc[2] = {vzero8(), vzero8()};

    for (int kb = 0; kb < D_DIM; kb += 32) {
        load_tile_64x32(aTile, a  + (size_t)blockM * D_DIM + kb, D_DIM, tid);
        load_tile_64x32(bTile, wb + (size_t)blockN * D_DIM + kb, D_DIM, tid);
        if (kb + 32 < D_DIM)
            __builtin_prefetch(a + (size_t)blockM * D_DIM + kb + 32, 0, 0);
        async_join_barrier();

        bf16x16 aLo = load_a_frag(aTile + subM * 32, 32, lane);
        bf16x16 aHi = load_a_frag(aTile + (subM + 16) * 32, 32, lane);
        bf16x16 bF  = load_b_frag(bTile + subN * 32, 32, lane);
        acc[0] = wmma_bf16(aLo, bF, acc[0]);
        acc[1] = wmma_bf16(aHi, bF, acc[1]);
        __syncthreads();
    }

    int n = blockN + subN + (lane & 15);
    float bias = bo[n];
    #pragma unroll
    for (int h2 = 0; h2 < 2; ++h2) {
        int mbase = blockM + subM + h2 * 16 + 8 * (lane >> 4);
        #pragma unroll
        for (int r = 0; r < 8; ++r) {
            int mrow = mbase + r;
            out[(size_t)mrow * D_DIM + n] = acc[h2][r] + bias;
        }
    }
}

// ---------------------------------------------------------------------------
// Host launch
// ---------------------------------------------------------------------------
extern "C" void kernel_launch(void* const* d_in, const int* in_sizes, int n_in,
                              void* d_out, int out_size, void* d_ws, size_t ws_size,
                              hipStream_t stream) {
    (void)in_sizes; (void)n_in; (void)out_size; (void)ws_size;

    const float*         m       = (const float*)d_in[0];
    const unsigned char* seq_pad = (const unsigned char*)d_in[1];
    const unsigned char* res_pad = (const unsigned char*)d_in[2];
    const float*         ln_g    = (const float*)d_in[3];
    const float*         ln_b    = (const float*)d_in[4];
    const float*         Wq      = (const float*)d_in[5];
    const float*         Wk      = (const float*)d_in[6];
    const float*         Wv      = (const float*)d_in[7];
    const float*         Wg      = (const float*)d_in[8];
    const float*         bg      = (const float*)d_in[9];
    const float*         Wo      = (const float*)d_in[10];
    const float*         bo      = (const float*)d_in[11];
    float*               out     = (float*)d_out;

    // Workspace carve-up (bf16 buffers)
    char* ws = (char*)d_ws;
    const size_t actN = (size_t)M_TOTAL * D_DIM;     // 8,388,608 elements
    const size_t wN   = (size_t)D_DIM * D_DIM;       // 65,536 elements

    bf16* mn   = (bf16*)ws;                 ws += actN * sizeof(bf16);
    bf16* wpk  = (bf16*)ws;                 ws += 5 * wN * sizeof(bf16);
    bf16* q_ws = (bf16*)ws;                 ws += actN * sizeof(bf16);
    bf16* k_ws = (bf16*)ws;                 ws += actN * sizeof(bf16);
    bf16* v_ws = (bf16*)ws;                 ws += actN * sizeof(bf16);
    bf16* g_ws = (bf16*)ws;                 ws += actN * sizeof(bf16);
    bf16* o_ws = (bf16*)ws;                 ws += actN * sizeof(bf16);

    bf16* wqb = wpk + 0 * wN;
    bf16* wkb = wpk + 1 * wN;
    bf16* wvb = wpk + 2 * wN;
    bf16* wgb = wpk + 3 * wN;
    bf16* wob = wpk + 4 * wN;

    // 1) LayerNorm -> bf16
    msa_ln_kernel<<<M_TOTAL / 8, NTHREADS, 0, stream>>>(m, ln_g, ln_b, mn);

    // 2) Weight conversion
    msa_cvtw_kernel<<<(5 * (int)wN) / NTHREADS, NTHREADS, 0, stream>>>(
        Wq, Wk, Wv, Wg, Wo, wpk);

    // 3) Fused QKVG projection
    msa_qkvg_kernel<<<dim3(M_TOTAL / 64, D_DIM / 64), NTHREADS, 0, stream>>>(
        mn, wqb, wkb, wvb, wgb, bg, q_ws, k_ws, v_ws, g_ws);

    // 4) Attention per (s,h)
    msa_attn_kernel<<<S_DIM * H_DIM, NTHREADS, 0, stream>>>(
        q_ws, k_ws, v_ws, g_ws, seq_pad, res_pad, o_ws);

    // 5) Output projection
    msa_oproj_kernel<<<dim3(M_TOTAL / 64, D_DIM / 64), NTHREADS, 0, stream>>>(
        o_ws, wob, bo, out);
}
